// GATSchemaLinker_85066122265470
// MI455X (gfx1250) — compile-verified, matrix operand order checked by reading, not run on hardware
//
#include <hip/hip_runtime.h>
#include <hip/hip_bf16.h>
#include <math.h>

typedef __attribute__((ext_vector_type(16))) _Float16 v16h;
typedef __attribute__((ext_vector_type(8)))  _Float16 v8h;
typedef __attribute__((ext_vector_type(8)))  float    v8f;

#define NEG_SLOPE 0.2f

__device__ __forceinline__ int f2oi(float f) {
  int i = __float_as_int(f);
  return i >= 0 ? i : (i ^ 0x7FFFFFFF);
}
__device__ __forceinline__ float oi2f(int i) {
  return __int_as_float(i >= 0 ? i : (i ^ 0x7FFFFFFF));
}
__device__ __forceinline__ float leaky(float x) { return x > 0.f ? x : NEG_SLOPE * x; }
__device__ __forceinline__ void atomAddF(float* p, float v) {
  __hip_atomic_fetch_add(p, v, __ATOMIC_RELAXED, __HIP_MEMORY_SCOPE_AGENT);
}

// ---------------- casts ----------------
__global__ void k_cast_f16(const float* __restrict__ in, _Float16* __restrict__ out, int n) {
  int t = blockIdx.x * blockDim.x + threadIdx.x;
  if (t < n) out[t] = (_Float16)in[t];
}

// in: [R,C] row-major f32 -> out: [C,R] row-major f16 (B^T so WMMA B loads are contiguous in K)
__global__ void k_castT_f16(const float* __restrict__ in, _Float16* __restrict__ out, int R, int C) {
  int t = blockIdx.x * blockDim.x + threadIdx.x;
  if (t >= R * C) return;
  int r = t / C, c = t - r * C;
  out[c * R + r] = (_Float16)in[t];
}

// ---------------- WMMA GEMM ----------------
// C[M,N] (f32, row-major) = A[M,K] (f16 row-major) x B[K,N] given as Bt[N,K] (f16 row-major).
// One wave per 16x64 strip: 4 accumulators share one A fetch per 32-K chunk (4x A reuse).
// Requires M%16==0, N%64==0, K%32==0 (true here: M=50000, N in {256,64}, K in {128,256}).
// 16-bit A/B lane layout per CDNA5 ISA: lanes 0-15 hold K {0..7,16..23},
// lanes 16-31 hold K {8..15,24..31} as packed pairs across 8 VGPRs.
__global__ void k_wmma_gemm(const _Float16* __restrict__ A,
                            const _Float16* __restrict__ Bt,
                            float* __restrict__ C,
                            int M, int N, int K) {
  const int tilesN      = N >> 4;
  const int wavesPerRow = tilesN >> 2;               // 4 col-tiles per wave
  const int totalWaves  = (M >> 4) * wavesPerRow;
  const int wave = blockIdx.x * (blockDim.x >> 5) + (threadIdx.x >> 5);
  if (wave >= totalWaves) return;
  const int lane = threadIdx.x & 31;
  const int half = lane >> 4;   // 0: K 0..7/16..23, 1: K 8..15/24..31
  const int l16  = lane & 15;
  const int tm  = wave / wavesPerRow;
  const int tn0 = (wave - tm * wavesPerRow) * 4;

  const _Float16* aRow = A + (size_t)(tm * 16 + l16) * K;
  const _Float16* bRow[4];
  #pragma unroll
  for (int j = 0; j < 4; ++j)
    bRow[j] = Bt + (size_t)((tn0 + j) * 16 + l16) * K;

  v8f acc0 = {}, acc1 = {}, acc2 = {}, acc3 = {};
  for (int k = 0; k < K; k += 32) {
    const int ka = k + half * 8;
    union { v16h v; v8h h[2]; } a, b0, b1, b2, b3;
    a.h[0]  = *(const v8h*)(aRow + ka);
    a.h[1]  = *(const v8h*)(aRow + ka + 16);
    b0.h[0] = *(const v8h*)(bRow[0] + ka);
    b0.h[1] = *(const v8h*)(bRow[0] + ka + 16);
    b1.h[0] = *(const v8h*)(bRow[1] + ka);
    b1.h[1] = *(const v8h*)(bRow[1] + ka + 16);
    b2.h[0] = *(const v8h*)(bRow[2] + ka);
    b2.h[1] = *(const v8h*)(bRow[2] + ka + 16);
    b3.h[0] = *(const v8h*)(bRow[3] + ka);
    b3.h[1] = *(const v8h*)(bRow[3] + ka + 16);
    acc0 = __builtin_amdgcn_wmma_f32_16x16x32_f16(false, a.v, false, b0.v, (short)0, acc0, false, false);
    acc1 = __builtin_amdgcn_wmma_f32_16x16x32_f16(false, a.v, false, b1.v, (short)0, acc1, false, false);
    acc2 = __builtin_amdgcn_wmma_f32_16x16x32_f16(false, a.v, false, b2.v, (short)0, acc2, false, false);
    acc3 = __builtin_amdgcn_wmma_f32_16x16x32_f16(false, a.v, false, b3.v, (short)0, acc3, false, false);
  }
  // C/D layout: VGPR j -> row (half*8 + j), col = l16
  float* cRow = C + (size_t)(tm * 16 + half * 8) * N + tn0 * 16 + l16;
  #pragma unroll
  for (int j = 0; j < 8; ++j) {
    float* cr = cRow + (size_t)j * N;
    cr[0]  = acc0[j];
    cr[16] = acc1[j];
    cr[32] = acc2[j];
    cr[48] = acc3[j];
  }
}

// ---------------- attention coefficients ----------------
// as_[n,h] = sum_c h[n,h,c]*att_s[h,c] ; ad_ likewise
__global__ void k_attn(const float* __restrict__ h, const float* __restrict__ att_s,
                       const float* __restrict__ att_d,
                       float* __restrict__ as_, float* __restrict__ ad_,
                       int Nn, int H, int Cc) {
  int t = blockIdx.x * blockDim.x + threadIdx.x;
  if (t >= Nn * H) return;
  int n = t / H, hh = t - n * H;
  const float* hp = h + (size_t)n * H * Cc + (size_t)hh * Cc;
  const float* sp = att_s + hh * Cc;
  const float* dp = att_d + hh * Cc;
  float a0 = 0.f, a1 = 0.f;
  for (int c = 0; c < Cc; ++c) { float v = hp[c]; a0 += v * sp[c]; a1 += v * dp[c]; }
  as_[t] = a0;
  ad_[t] = a1;
}

// ---------------- init (zero accumulator, -inf segment max, zero segment sum) ----------------
__global__ void k_init(float* __restrict__ agg, int nAgg,
                       int* __restrict__ m, float* __restrict__ s, int nMS) {
  int t = blockIdx.x * blockDim.x + threadIdx.x;
  if (t < nAgg) agg[t] = 0.f;
  if (t < nMS) { m[t] = f2oi(-__builtin_inff()); s[t] = 0.f; }
}

// ---------------- edge passes ----------------
__global__ void k_edge_max(const int* __restrict__ src, const int* __restrict__ dst,
                           int E, int Etot,
                           const float* __restrict__ as_, const float* __restrict__ ad_,
                           int H, int* __restrict__ m) {
  int t = blockIdx.x * blockDim.x + threadIdx.x;
  if (t >= Etot * H) return;
  int e = t / H, hh = t - e * H;
  int s_ = e < E ? src[e] : (e - E);   // self loops appended after real edges
  int d_ = e < E ? dst[e] : (e - E);
  float ev = leaky(as_[s_ * H + hh] + ad_[d_ * H + hh]);
  atomicMax(&m[d_ * H + hh], f2oi(ev));
}

__global__ void k_edge_sum(const int* __restrict__ src, const int* __restrict__ dst,
                           int E, int Etot,
                           const float* __restrict__ as_, const float* __restrict__ ad_,
                           int H, const int* __restrict__ m, float* __restrict__ s) {
  int t = blockIdx.x * blockDim.x + threadIdx.x;
  if (t >= Etot * H) return;
  int e = t / H, hh = t - e * H;
  int s_ = e < E ? src[e] : (e - E);
  int d_ = e < E ? dst[e] : (e - E);
  float ev = leaky(as_[s_ * H + hh] + ad_[d_ * H + hh]);
  atomAddF(&s[d_ * H + hh], expf(ev - oi2f(m[d_ * H + hh])));
}

// one wave per edge; alphas hoisted (wave-uniform), lanes sweep H*C in 32-wide coalesced chunks
__global__ void k_edge_agg(const int* __restrict__ src, const int* __restrict__ dst,
                           int E, int Etot,
                           const float* __restrict__ as_, const float* __restrict__ ad_,
                           const int* __restrict__ m, const float* __restrict__ ssum,
                           const float* __restrict__ feat, float* __restrict__ outp,
                           int H, int Cc) {
  int wid = blockIdx.x * (blockDim.x >> 5) + (threadIdx.x >> 5);
  if (wid >= Etot) return;
  int lane = threadIdx.x & 31;
  int s_ = wid < E ? src[wid] : (wid - E);
  int d_ = wid < E ? dst[wid] : (wid - E);
  int HC = H * Cc;
  const float* fRow = feat + (size_t)s_ * HC;
  float*       oRow = outp + (size_t)d_ * HC;

  // get the gathered feature row in flight early (global_prefetch)
  __builtin_prefetch(&fRow[(size_t)lane * HC / 32], 0, 0);

  float alpha[4];
  for (int hh = 0; hh < H; ++hh) {
    float ev = leaky(as_[s_ * H + hh] + ad_[d_ * H + hh]);
    alpha[hh] = expf(ev - oi2f(m[d_ * H + hh])) / (ssum[d_ * H + hh] + 1e-16f);
  }
  for (int base = 0; base < HC; base += 32) {
    float al = alpha[base / Cc];   // Cc multiple of 32 -> uniform within chunk
    int idx = base + lane;
    atomAddF(&oRow[idx], fRow[idx] * al);
  }
}

// ---------------- epilogues ----------------
__global__ void k_bias_relu_cast(const float* __restrict__ a, const float* __restrict__ b,
                                 _Float16* __restrict__ o, int n, int hcMask) {
  int t = blockIdx.x * blockDim.x + threadIdx.x;
  if (t >= n) return;
  float v = a[t] + b[t & hcMask];
  o[t] = (_Float16)(v > 0.f ? v : 0.f);
}

__global__ void k_bias_add(float* __restrict__ o, const float* __restrict__ b, int n, int hcMask) {
  int t = blockIdx.x * blockDim.x + threadIdx.x;
  if (t >= n) return;
  o[t] += b[t & hcMask];
}

// ---------------- host launcher ----------------
extern "C" void kernel_launch(void* const* d_in, const int* in_sizes, int n_in,
                              void* d_out, int out_size, void* d_ws, size_t ws_size,
                              hipStream_t stream) {
  const int IN = 128, HID = 64, H1 = 4;
  const int HC1 = H1 * HID;  // 256
  const int HC2 = HID;       // 64
  const int Nn   = in_sizes[0] / IN;
  const int E    = in_sizes[1] / 2;
  const int Etot = E + Nn;

  const float* x   = (const float*)d_in[0];
  const int*   ei  = (const int*)d_in[1];
  const float* W1  = (const float*)d_in[2];
  const float* aS1 = (const float*)d_in[3];
  const float* aD1 = (const float*)d_in[4];
  const float* b1  = (const float*)d_in[5];
  const float* W2  = (const float*)d_in[6];
  const float* aS2 = (const float*)d_in[7];
  const float* aD2 = (const float*)d_in[8];
  const float* b2  = (const float*)d_in[9];
  float* out = (float*)d_out;
  const int* srcI = ei;
  const int* dstI = ei + E;

  // carve workspace (256B aligned chunks)
  char* p = (char*)d_ws;
  auto carve = [&](size_t bytes) -> void* {
    void* r = (void*)p;
    p += (bytes + 255) & ~(size_t)255;
    return r;
  };
  _Float16* xh   = (_Float16*)carve((size_t)Nn * IN * 2);
  _Float16* w1t  = (_Float16*)carve((size_t)HC1 * IN * 2);
  _Float16* w2t  = (_Float16*)carve((size_t)HID * HC1 * 2);
  float*    h1   = (float*)carve((size_t)Nn * HC1 * 4);
  float*    agg1 = (float*)carve((size_t)Nn * HC1 * 4);
  _Float16* h1h  = (_Float16*)carve((size_t)Nn * HC1 * 2);
  float*    h2   = (float*)carve((size_t)Nn * HC2 * 4);
  float*    as1  = (float*)carve((size_t)Nn * H1 * 4);
  float*    ad1  = (float*)carve((size_t)Nn * H1 * 4);
  int*      m1   = (int*)carve((size_t)Nn * H1 * 4);
  float*    s1   = (float*)carve((size_t)Nn * H1 * 4);
  float*    as2  = (float*)carve((size_t)Nn * 4);
  float*    ad2  = (float*)carve((size_t)Nn * 4);
  int*      m2   = (int*)carve((size_t)Nn * 4);
  float*    s2   = (float*)carve((size_t)Nn * 4);

  const int TB = 256;
  auto cdiv = [](long long a, long long b) { return (int)((a + b - 1) / b); };

  // precision casts / weight transposes
  k_cast_f16<<<cdiv((long long)Nn * IN, TB), TB, 0, stream>>>(x, xh, Nn * IN);
  k_castT_f16<<<cdiv((long long)IN * HC1, TB), TB, 0, stream>>>(W1, w1t, IN, HC1);
  k_castT_f16<<<cdiv((long long)HC1 * HID, TB), TB, 0, stream>>>(W2, w2t, HC1, HID);

  // ---- layer 1 ----
  {
    int waves = (Nn / 16) * (HC1 / 64);
    k_wmma_gemm<<<cdiv(waves, 8), TB, 0, stream>>>(xh, w1t, h1, Nn, HC1, IN);
  }
  k_attn<<<cdiv((long long)Nn * H1, TB), TB, 0, stream>>>(h1, aS1, aD1, as1, ad1, Nn, H1, HID);
  k_init<<<cdiv((long long)Nn * HC1, TB), TB, 0, stream>>>(agg1, Nn * HC1, m1, s1, Nn * H1);
  k_edge_max<<<cdiv((long long)Etot * H1, TB), TB, 0, stream>>>(srcI, dstI, E, Etot, as1, ad1, H1, m1);
  k_edge_sum<<<cdiv((long long)Etot * H1, TB), TB, 0, stream>>>(srcI, dstI, E, Etot, as1, ad1, H1, m1, s1);
  k_edge_agg<<<cdiv(Etot, 8), TB, 0, stream>>>(srcI, dstI, E, Etot, as1, ad1, m1, s1, h1, agg1, H1, HID);
  k_bias_relu_cast<<<cdiv((long long)Nn * HC1, TB), TB, 0, stream>>>(agg1, b1, h1h, Nn * HC1, HC1 - 1);

  // ---- layer 2 ----
  {
    int waves = (Nn / 16) * (HC2 / 64);
    k_wmma_gemm<<<cdiv(waves, 8), TB, 0, stream>>>(h1h, w2t, h2, Nn, HC2, HC1);
  }
  k_attn<<<cdiv((long long)Nn, TB), TB, 0, stream>>>(h2, aS2, aD2, as2, ad2, Nn, 1, HID);
  k_init<<<cdiv((long long)Nn * HC2, TB), TB, 0, stream>>>(out, Nn * HC2, m2, s2, Nn);
  k_edge_max<<<cdiv((long long)Etot, TB), TB, 0, stream>>>(srcI, dstI, E, Etot, as2, ad2, 1, m2);
  k_edge_sum<<<cdiv((long long)Etot, TB), TB, 0, stream>>>(srcI, dstI, E, Etot, as2, ad2, 1, m2, s2);
  k_edge_agg<<<cdiv(Etot, 8), TB, 0, stream>>>(srcI, dstI, E, Etot, as2, ad2, m2, s2, h2, out, 1, HID);
  k_bias_add<<<cdiv((long long)Nn * HC2, TB), TB, 0, stream>>>(out, b2, Nn * HC2, HC2 - 1);
}
